// Head_58093727646031
// MI455X (gfx1250) — compile-verified, matrix operand order checked by reading, not run on hardware
//
#include <hip/hip_runtime.h>

#define B_ 8
#define T_ 2048
#define C_ 1024
#define H_ 64

typedef __attribute__((ext_vector_type(16))) _Float16 v16h;
typedef __attribute__((ext_vector_type(8)))  _Float16 v8h;
typedef __attribute__((ext_vector_type(8)))  float    v8f;

#define WMMA_F16(A, Bm, Cm) \
  __builtin_amdgcn_wmma_f32_16x16x32_f16(false, (A), false, (Bm), (short)0, (Cm), false, false)

// 16-half fragment from a contiguous f16 row:
// elems 0..7 <- row[b0..b0+7], elems 8..15 <- row[b1..b1+7]
// (A-layout row-per-lane and B-layout col-per-lane use the same per-lane mapping)
static __device__ __forceinline__ v16h frag_from_f16_row(const _Float16* row, int b0, int b1) {
  v8h lo = *(const v8h*)(row + b0);
  v8h hi = *(const v8h*)(row + b1);
  return __builtin_shufflevector(lo, hi, 0,1,2,3,4,5,6,7,8,9,10,11,12,13,14,15);
}

// Same but from f32 (converted on the fly)
static __device__ __forceinline__ v16h frag_from_f32_row(const float* row, int b0, int b1) {
  v16h r;
#pragma unroll
  for (int e = 0; e < 8; ++e) r[e]     = (_Float16)row[b0 + e];
#pragma unroll
  for (int e = 0; e < 8; ++e) r[8 + e] = (_Float16)row[b1 + e];
  return r;
}

// ---------------------------------------------------------------------------
// Kernel 0: one-time W transpose+downconvert: W[1024][64] f32 -> wt[64][1024] f16
// ---------------------------------------------------------------------------
__global__ __launch_bounds__(256) void wt_kernel(
    const float* __restrict__ Wq, const float* __restrict__ Wk,
    const float* __restrict__ Wv, _Float16* __restrict__ wt)
{
  const int idx = blockIdx.x * 256 + threadIdx.x;   // 0 .. 3*64*1024-1
  const int n   = idx & 63;                          // coalesced reads over n
  const int kk  = (idx >> 6) & 1023;
  const int mat = idx >> 16;
  const float* W = (mat == 0) ? Wq : (mat == 1) ? Wk : Wv;
  wt[((size_t)mat * H_ + n) * C_ + kk] = (_Float16)W[(size_t)kk * H_ + n];
}

// ---------------------------------------------------------------------------
// Kernel 1: projections. One wave per (16-row tile, matrix); the x fragment is
// loaded once per K-step and reused by 4 WMMAs.  The wave-uniform mat==2 role
// swap is hoisted OUT of the K-loop so each inner loop is straight-line.
//   q,k : D = x · W    -> row-major f16 [B*T][64]
//   v   : D = Wᵀ · xᵀ  -> TRANSPOSED f16 [B][64][2048] (coalesced stores, and
//                         exactly the layout attention's Vᵀ A-frags want)
// ---------------------------------------------------------------------------
__global__ __launch_bounds__(256) void proj_kernel(
    const float* __restrict__ x, const _Float16* __restrict__ wt,
    _Float16* __restrict__ q, _Float16* __restrict__ k, _Float16* __restrict__ vT)
{
  const int wid  = (blockIdx.x * blockDim.x + threadIdx.x) >> 5;
  const int lane = threadIdx.x & 31;
  const int half = lane >> 4;
  const int l16  = lane & 15;

  const int mat = wid % 3;          // 0:q 1:k 2:v
  const int rt  = wid / 3;          // 16-row tile of B*T (0..1023)

  const float*    xrow = x + (size_t)(rt * 16 + l16) * C_;
  const _Float16* wtm  = wt + (size_t)mat * H_ * C_;
  // One wt row per lane per n-tile; serves as B-col (q,k) or A-row (v).
  const _Float16* wr0 = wtm + (size_t)(0 * 16 + l16) * C_;
  const _Float16* wr1 = wtm + (size_t)(1 * 16 + l16) * C_;
  const _Float16* wr2 = wtm + (size_t)(2 * 16 + l16) * C_;
  const _Float16* wr3 = wtm + (size_t)(3 * 16 + l16) * C_;

  v8f a0 = {}, a1 = {}, a2 = {}, a3 = {};

  if (mat == 2) {                       // vᵀ = Wᵀ·xᵀ : W rows are the A matrix
    for (int kk = 0; kk < C_; kk += 32) {
      const int b0 = kk + half * 8, b1 = kk + 16 + half * 8;
      v16h xf = frag_from_f32_row(xrow, b0, b1);
      a0 = WMMA_F16(frag_from_f16_row(wr0, b0, b1), xf, a0);
      a1 = WMMA_F16(frag_from_f16_row(wr1, b0, b1), xf, a1);
      a2 = WMMA_F16(frag_from_f16_row(wr2, b0, b1), xf, a2);
      a3 = WMMA_F16(frag_from_f16_row(wr3, b0, b1), xf, a3);
    }
    // C/D: row = h (nt*16 + vg + 8*half), col = token s (lane) -> coalesced
    const int b  = rt >> 7;
    const int s0 = (rt & 127) * 16;
    _Float16* vtb = vT + (size_t)b * H_ * T_ + s0 + l16;
#pragma unroll
    for (int vg = 0; vg < 8; ++vg) {
      const int hr = vg + half * 8;
      vtb[(size_t)(0 * 16 + hr) * T_] = (_Float16)a0[vg];
      vtb[(size_t)(1 * 16 + hr) * T_] = (_Float16)a1[vg];
      vtb[(size_t)(2 * 16 + hr) * T_] = (_Float16)a2[vg];
      vtb[(size_t)(3 * 16 + hr) * T_] = (_Float16)a3[vg];
    }
  } else {                              // q,k = x·W : x rows are the A matrix
    for (int kk = 0; kk < C_; kk += 32) {
      const int b0 = kk + half * 8, b1 = kk + 16 + half * 8;
      v16h xf = frag_from_f32_row(xrow, b0, b1);
      a0 = WMMA_F16(xf, frag_from_f16_row(wr0, b0, b1), a0);
      a1 = WMMA_F16(xf, frag_from_f16_row(wr1, b0, b1), a1);
      a2 = WMMA_F16(xf, frag_from_f16_row(wr2, b0, b1), a2);
      a3 = WMMA_F16(xf, frag_from_f16_row(wr3, b0, b1), a3);
    }
    // C/D: row = token (vg + 8*half), col = h (nt*16 + lane)
    _Float16* orow = ((mat == 0) ? q : k) + (size_t)(rt * 16) * H_ + l16;
#pragma unroll
    for (int vg = 0; vg < 8; ++vg) {
      _Float16* p = orow + (size_t)(vg + half * 8) * H_;
      p[0 * 16] = (_Float16)a0[vg];
      p[1 * 16] = (_Float16)a1[vg];
      p[2 * 16] = (_Float16)a2[vg];
      p[3 * 16] = (_Float16)a3[vg];
    }
  }
}

// ---------------------------------------------------------------------------
// Kernel 2: causal flash attention. One wave per 16-query tile.
//   Sᵀ = K·Qᵀ  -> each lane owns one query row; stats need one shfl_xor(16)
//   Oᵀ = Vᵀ·Pᵀ -> Pᵀ B-frag is a pure per-lane f32->f16 pack; Vᵀ A-frags are
//                 contiguous 16B loads thanks to the transposed V layout.
// ---------------------------------------------------------------------------
__global__ __launch_bounds__(256) void attn_kernel(
    const _Float16* __restrict__ q, const _Float16* __restrict__ kmat,
    const _Float16* __restrict__ vT, float* __restrict__ out)
{
  const int wid  = (blockIdx.x * blockDim.x + threadIdx.x) >> 5;
  const int lane = threadIdx.x & 31;
  const int half = lane >> 4;
  const int l16  = lane & 15;

  const int b  = wid >> 7;
  const int t0 = (wid & 127) * 16;
  const int t  = t0 + l16;                       // this lane's query row

  const _Float16* qb  = q    + (size_t)b * T_ * H_;
  const _Float16* kb  = kmat + (size_t)b * T_ * H_;
  const _Float16* vtb = vT   + (size_t)b * H_ * T_;

  const _Float16* qrow = qb + (size_t)t * H_;
  const v16h qf0 = frag_from_f16_row(qrow,      half * 8, 16 + half * 8);
  const v16h qf1 = frag_from_f16_row(qrow, 32 + half * 8, 48 + half * 8);

  // Vᵀ rows this lane reads (row h per lane, elems = keys, contiguous)
  const _Float16* vr0 = vtb + (size_t)(0 * 16 + l16) * T_;
  const _Float16* vr1 = vtb + (size_t)(1 * 16 + l16) * T_;
  const _Float16* vr2 = vtb + (size_t)(2 * 16 + l16) * T_;
  const _Float16* vr3 = vtb + (size_t)(3 * 16 + l16) * T_;

  v8f o0 = {}, o1 = {}, o2 = {}, o3 = {};
  float m_i = -1e30f, l_i = 0.0f;
  const float scale = 0.125f;                    // 1/sqrt(64)

  for (int s_base = 0; s_base < t0 + 16; s_base += 32) {
    // K A-fragments (row = key per lane, elems = h, contiguous)
    const _Float16* kr0 = kb + (size_t)(s_base +      l16) * H_;
    const _Float16* kr1 = kb + (size_t)(s_base + 16 + l16) * H_;
    v16h ka00 = frag_from_f16_row(kr0,      half * 8, 16 + half * 8);
    v16h ka01 = frag_from_f16_row(kr0, 32 + half * 8, 48 + half * 8);
    v16h ka10 = frag_from_f16_row(kr1,      half * 8, 16 + half * 8);
    v16h ka11 = frag_from_f16_row(kr1, 32 + half * 8, 48 + half * 8);

    v8f s0 = {}, s1 = {};
    s0 = WMMA_F16(ka00, qf0, s0);  s0 = WMMA_F16(ka01, qf1, s0);
    s1 = WMMA_F16(ka10, qf0, s1);  s1 = WMMA_F16(ka11, qf1, s1);

    // scale + causal mask + chunk max
    float cmax = -1e30f;
#pragma unroll
    for (int i = 0; i < 8; ++i) {
      const int key0 = s_base +      half * 8 + i;
      const int key1 = s_base + 16 + half * 8 + i;
      float e0 = (key0 <= t) ? s0[i] * scale : -1e30f;
      float e1 = (key1 <= t) ? s1[i] * scale : -1e30f;
      s0[i] = e0;  s1[i] = e1;
      cmax = fmaxf(cmax, fmaxf(e0, e1));
    }
    cmax = fmaxf(cmax, __shfl_xor(cmax, 16, 32));

    const float m_new = fmaxf(m_i, cmax);
    const float alpha = __expf(m_i - m_new);

    float rsum = 0.0f;
    v16h pf;                                     // Pᵀ B-fragment (per-lane pack)
#pragma unroll
    for (int i = 0; i < 8; ++i) {
      const float p0 = __expf(s0[i] - m_new);
      const float p1 = __expf(s1[i] - m_new);
      rsum += p0 + p1;
      pf[i]     = (_Float16)p0;
      pf[8 + i] = (_Float16)p1;
    }
    rsum += __shfl_xor(rsum, 16, 32);
    l_i = l_i * alpha + rsum;
    m_i = m_new;

#pragma unroll
    for (int i = 0; i < 8; ++i) { o0[i] *= alpha; o1[i] *= alpha; o2[i] *= alpha; o3[i] *= alpha; }

    // Oᵀ += Vᵀ·Pᵀ — Vᵀ A-frags contiguous
    const int b0 = s_base + half * 8, b1 = s_base + 16 + half * 8;
    o0 = WMMA_F16(frag_from_f16_row(vr0, b0, b1), pf, o0);
    o1 = WMMA_F16(frag_from_f16_row(vr1, b0, b1), pf, o1);
    o2 = WMMA_F16(frag_from_f16_row(vr2, b0, b1), pf, o2);
    o3 = WMMA_F16(frag_from_f16_row(vr3, b0, b1), pf, o3);
  }

  // Oᵀ: col = query (lane), row h = nt*16 + vg + 8*half
  const float inv = 1.0f / l_i;
  float* orow = out + ((size_t)b * T_ + t) * H_;
#pragma unroll
  for (int vg = 0; vg < 8; ++vg) {
    orow[0 * 16 + vg + half * 8] = o0[vg] * inv;
    orow[1 * 16 + vg + half * 8] = o1[vg] * inv;
    orow[2 * 16 + vg + half * 8] = o2[vg] * inv;
    orow[3 * 16 + vg + half * 8] = o3[vg] * inv;
  }
}

extern "C" void kernel_launch(void* const* d_in, const int* in_sizes, int n_in,
                              void* d_out, int out_size, void* d_ws, size_t ws_size,
                              hipStream_t stream) {
  const float* x  = (const float*)d_in[0];
  const float* Wq = (const float*)d_in[1];
  const float* Wk = (const float*)d_in[2];
  const float* Wv = (const float*)d_in[3];
  float* out = (float*)d_out;

  _Float16* qws  = (_Float16*)d_ws;
  _Float16* kws  = qws + (size_t)B_ * T_ * H_;
  _Float16* vtws = kws + (size_t)B_ * T_ * H_;
  _Float16* wtws = vtws + (size_t)B_ * T_ * H_;   // 3*64*1024 f16

  // W transpose: 3*64*1024 elements, one per thread
  wt_kernel<<<(3 * H_ * C_) / 256, 256, 0, stream>>>(Wq, Wk, Wv, wtws);

  // Projections: 1024 row tiles * 3 matrices = 3072 waves, 8 waves/block
  proj_kernel<<<(B_ * T_ / 16) * 3 / 8, 256, 0, stream>>>(x, wtws, qws, kws, vtws);

  // Attention: one wave per 16-query tile = 1024 waves
  attn_kernel<<<(B_ * T_ / 16) / 8, 256, 0, stream>>>(qws, kws, vtws, out);
}
